// DeChunkLayer_88304527606428
// MI455X (gfx1250) — matrix operand order repaired and checked by reading, not compiled, not run on hardware
//
#include <hip/hip_runtime.h>
#include <hip/hip_bf16.h>
#include <math.h>

typedef __attribute__((ext_vector_type(16))) _Float16 v16h;
typedef __attribute__((ext_vector_type(8)))  float    v8f;

#define BATCH  2
#define SEQ    4096
#define DMODEL 1024
#define CH     64            // our chunk length (math identical to reference's 256)
#define NC     (SEQ / CH)    // 64 chunks per batch
#define DN     64            // channel tile per workgroup (4 waves x 16 cols)
#define LPAD   (CH + 2)      // LDS row pad: stride 66 halves -> conflict-free row gathers

// ---------------------------------------------------------------------------
// Phase A: stable-partition gather of p, dt/w, chunk-local cumsum s, plug-back idx
// One 1024-thread block per batch; Hillis-Steele scans in LDS.
// ---------------------------------------------------------------------------
__global__ __launch_bounds__(1024)
void prep_kernel(const float* __restrict__ bprob,
                 const unsigned char* __restrict__ bmask,   // numpy bool = 1 byte
                 float* __restrict__ s_ws,
                 float* __restrict__ w_ws,
                 int*   __restrict__ pbi_ws)
{
    __shared__ int   si[SEQ];
    __shared__ float ps[SEQ];
    __shared__ float sf[SEQ];
    const int b   = blockIdx.x;
    const int tid = threadIdx.x;

    int m[4];
    #pragma unroll
    for (int k = 0; k < 4; k++) {
        int i = tid + k * 1024;
        m[k] = bmask[b * SEQ + i] ? 1 : 0;
        si[i] = m[k];
    }
    __syncthreads();
    // inclusive scan of mask over 4096
    for (int d = 1; d < SEQ; d <<= 1) {
        int a[4];
        #pragma unroll
        for (int k = 0; k < 4; k++) {
            int i = tid + k * 1024;
            a[k] = si[i] + ((i >= d) ? si[i - d] : 0);
        }
        __syncthreads();
        #pragma unroll
        for (int k = 0; k < 4; k++) si[tid + k * 1024] = a[k];
        __syncthreads();
    }
    const int nT = si[SEQ - 1];
    #pragma unroll
    for (int k = 0; k < 4; k++) {
        int i    = tid + k * 1024;
        int incl = si[i];
        pbi_ws[b * SEQ + i] = incl - 1;               // cumsum(mask)-1 (mask[0]=True)
        int excl = incl - m[k];
        int dest = m[k] ? excl : (nT + i - excl);     // stable partition == argsort
        float p = bprob[(size_t)(b * SEQ + i) * 2 + 1];
        p = fminf(fmaxf(p, 1e-4f), 1.f - 1e-4f);
        ps[dest] = p;
    }
    __syncthreads();
    #pragma unroll
    for (int k = 0; k < 4; k++) {
        int i = tid + k * 1024;
        float p  = ps[i];
        float dt = -log1pf(-p);                        // log(1/(1-p))
        w_ws[b * SEQ + i] = p / dt;                    // O(1) folded weight
        sf[i] = dt;
    }
    __syncthreads();
    // segmented (chunk-local, 64) inclusive cumsum of dt
    for (int d = 1; d < CH; d <<= 1) {
        float a[4];
        #pragma unroll
        for (int k = 0; k < 4; k++) {
            int i = tid + k * 1024;
            a[k] = sf[i] + (((i & (CH - 1)) >= d) ? sf[i - d] : 0.f);
        }
        __syncthreads();
        #pragma unroll
        for (int k = 0; k < 4; k++) sf[tid + k * 1024] = a[k];
        __syncthreads();
    }
    #pragma unroll
    for (int k = 0; k < 4; k++) {
        int i = tid + k * 1024;
        s_ws[b * SEQ + i] = sf[i];
    }
}

// Fragment index maps (CDNA5 ISA 7.12.2, wave32, 16-bit operands).
// A (16x32, lane = row M): halves 0..7 -> K=base+0..7, 8..15 -> K=base+16..23,
//   base = 0 (lanes 0-15) / 8 (lanes 16-31).
// B (32x16, lane = col N): same K mapping as A.
__device__ __forceinline__ int ka_map(int e, int lane) {
    return ((lane < 16) ? 0 : 8) + ((e < 8) ? e : e + 8);
}
__device__ __forceinline__ int kb_map(int e, int lane) {
    return ((lane < 16) ? 0 : 8) + ((e < 8) ? e : e + 8);
}

// ---------------------------------------------------------------------------
// Phase B: per (batch, chunk, 64-channel tile): Y_diag = L(64x64, lower-tri) @ Xp(64x64)
// via v_wmma_f32_16x16x32_f16, triangular tiles skipped.
// ---------------------------------------------------------------------------
__global__ __launch_bounds__(128)
void chunk_gemm_kernel(const float* __restrict__ x,
                       const float* __restrict__ s_ws,
                       const float* __restrict__ w_ws,
                       float* __restrict__ yd)
{
    __shared__ float    ss[CH];
    __shared__ float    sw[CH];
    __shared__ _Float16 Lm[CH][LPAD];
    __shared__ _Float16 Xp[CH][LPAD];

    const int b   = blockIdx.z;
    const int c   = blockIdx.y;
    const int n0  = blockIdx.x * DN;
    const int tid = threadIdx.x;

    if (tid < CH) {
        ss[tid] = s_ws[b * SEQ + c * CH + tid];
        sw[tid] = w_ws[b * SEQ + c * CH + tid];
    }
    __syncthreads();

    // decay matrix L[t][s] = exp(s_s - s_t), s<=t (exponent <= 0: underflow-safe)
    for (int idx = tid; idx < CH * CH; idx += 128) {
        int t = idx >> 6, s2 = idx & 63;
        float v = (s2 <= t) ? __expf(ss[s2] - ss[t]) : 0.f;
        Lm[t][s2] = (_Float16)v;
    }
    // Xp[s][n] = w_s * hidden[b, chunk_s, n]  (coalesced 64-float rows)
    for (int idx = tid; idx < CH * CH; idx += 128) {
        int sr = idx >> 6, n = idx & 63;
        Xp[sr][n] = (_Float16)(sw[sr] *
            x[((size_t)b * SEQ + (size_t)c * CH + sr) * DMODEL + n0 + n]);
    }
    __syncthreads();

    const int lane = tid & 31;
    const int nb   = (tid >> 5) * 16;       // this wave's 16-column slice
    const int nloc = nb + (lane & 15);

    v16h bf[2];
    #pragma unroll
    for (int kt = 0; kt < 2; kt++)
        #pragma unroll
        for (int e = 0; e < 16; e++)
            bf[kt][e] = Xp[kt * 32 + kb_map(e, lane)][nloc];

    #pragma unroll
    for (int mt = 0; mt < 4; mt++) {
        v8f acc = {};
        const int ktmax = (mt >= 2) ? 2 : 1;   // triangular: skip all-zero K tiles
        for (int kt = 0; kt < ktmax; kt++) {
            v16h a;
            #pragma unroll
            for (int e = 0; e < 16; e++)
                a[e] = Lm[mt * 16 + (lane & 15)][kt * 32 + ka_map(e, lane)];
            acc = __builtin_amdgcn_wmma_f32_16x16x32_f16(
                false, a, false, bf[kt], (short)0, acc, false, false);
        }
        #pragma unroll
        for (int v = 0; v < 8; v++) {
            int t = c * CH + mt * 16 + v + ((lane < 16) ? 0 : 8);  // C/D layout
            yd[((size_t)b * SEQ + t) * DMODEL + n0 + nloc] = acc[v];
        }
    }
}

// ---------------------------------------------------------------------------
// Phase C: cross-chunk carry.  Incoming state of chunk c == y at end of chunk c-1,
// and the chunk "state" S_c is exactly Y_diag's last row.  64 serial FMAs.
// ---------------------------------------------------------------------------
__global__ __launch_bounds__(256)
void state_scan_kernel(const float* __restrict__ yd,
                       const float* __restrict__ s_ws,
                       float* __restrict__ hws)
{
    int g = blockIdx.x * 256 + threadIdx.x;
    if (g >= BATCH * DMODEL) return;
    int b = g / DMODEL, d = g % DMODEL;
    float h = 0.f;
    for (int c = 0; c < NC; c++) {
        hws[((size_t)b * NC + c) * DMODEL + d] = h;     // incoming state for chunk c
        int tl  = c * CH + CH - 1;
        float T = s_ws[b * SEQ + tl];                   // full-chunk decay sum
        h = __expf(-T) * h + yd[((size_t)b * SEQ + tl) * DMODEL + d];
    }
}

// ---------------------------------------------------------------------------
// Phase D: out[b,i,:] = ( Y_diag[j] + exp(-s_j) * H_chunk(j) ),  j = plug_back_idx[i]
// ---------------------------------------------------------------------------
__global__ __launch_bounds__(256)
void combine_kernel(const float* __restrict__ yd,
                    const float* __restrict__ s_ws,
                    const float* __restrict__ hws,
                    const int*   __restrict__ pbi,
                    float* __restrict__ out)
{
    size_t g     = (size_t)blockIdx.x * 256 + threadIdx.x;
    size_t total = (size_t)BATCH * SEQ * DMODEL;
    if (g >= total) return;
    int    d = (int)(g % DMODEL);
    size_t r = g / DMODEL;
    int    i = (int)(r % SEQ);
    int    b = (int)(r / SEQ);
    int    j = pbi[b * SEQ + i];
    int    c = j / CH;
    float v = yd[((size_t)b * SEQ + j) * DMODEL + d] +
              __expf(-s_ws[b * SEQ + j]) * hws[((size_t)b * NC + c) * DMODEL + d];
    out[g] = v;
}

// ---------------------------------------------------------------------------
extern "C" void kernel_launch(void* const* d_in, const int* in_sizes, int n_in,
                              void* d_out, int out_size, void* d_ws, size_t ws_size,
                              hipStream_t stream)
{
    const float*         x     = (const float*)d_in[0];           // (2,4096,1024) f32
    const float*         bprob = (const float*)d_in[1];           // (2,4096,2)    f32
    const unsigned char* bmask = (const unsigned char*)d_in[2];   // (2,4096)      bool
    // d_in[3] (mask) is unused by the reference.

    float* s_ws  = (float*)d_ws;                      // chunk-local cumsum of dt
    float* w_ws  = s_ws + BATCH * SEQ;                // p/dt weights
    int*   pbi   = (int*)(w_ws + BATCH * SEQ);        // plug-back indices
    float* hws   = (float*)(pbi + BATCH * SEQ);       // incoming states, B*NC*D
    float* yd    = hws + (size_t)BATCH * NC * DMODEL; // Y_diag, B*L*D

    prep_kernel<<<BATCH, 1024, 0, stream>>>(bprob, bmask, s_ws, w_ws, pbi);

    chunk_gemm_kernel<<<dim3(DMODEL / DN, NC, BATCH), 128, 0, stream>>>(
        x, s_ws, w_ws, yd);

    state_scan_kernel<<<(BATCH * DMODEL + 255) / 256, 256, 0, stream>>>(
        yd, s_ws, hws);

    size_t total = (size_t)BATCH * SEQ * DMODEL;
    combine_kernel<<<(unsigned)((total + 255) / 256), 256, 0, stream>>>(
        yd, s_ws, hws, pbi, (float*)d_out);
}
